// PFN_38293928411471
// MI455X (gfx1250) — matrix operand order repaired and checked by reading, not compiled, not run on hardware
//
#include <hip/hip_runtime.h>
#include <hip/hip_bf16.h>

typedef __attribute__((ext_vector_type(16))) _Float16 v16h;
typedef __attribute__((ext_vector_type(8)))  float    v8f;
typedef __attribute__((ext_vector_type(4)))  float    v4f;
typedef _Float16 h16;

#define S_   512
#define B_   32
#define I_   768
#define H_   256
#define NH_  8
#define HD_  32
#define ML_  512
#define G5_  1280
#define CAT_ 768
#define SB_  (S_*B_)
#define LN_EPS 1e-5f

// ---------------------------------------------------------------------------
// WMMA helpers (CDNA5 gfx1250, wave32, V_WMMA_F32_16X16X32_F16)
// ---------------------------------------------------------------------------
__device__ __forceinline__ v8f wmma16(v16h a, v16h b, v8f c) {
  return __builtin_amdgcn_wmma_f32_16x16x32_f16(false, a, false, b, (short)0, c,
                                                false, false);
}

// A operand: 16x32 f16 tile from row-major src (row stride ld).
// lane L: row m = L%16; K=(L/16)*16+j (16 contiguous halves)
__device__ __forceinline__ v16h load_a16(const h16* __restrict__ src, int ld,
                                         int lm, int lhi) {
  v16h a;
  const h16* r = src + lm * ld + lhi * 16;
#pragma unroll
  for (int j = 0; j < 16; ++j) a[j] = r[j];
  return a;
}

// Same, but source is f32 (converted on the fly; used for softmax probs in LDS)
__device__ __forceinline__ v16h load_a32(const float* __restrict__ src, int ld,
                                         int lm, int lhi) {
  v16h a;
  const float* r = src + lm * ld + lhi * 16;
#pragma unroll
  for (int j = 0; j < 16; ++j) a[j] = (h16)r[j];
  return a;
}

// B operand for C = A @ W^T with row-major W[N][K]: B(k,n)=W[n][k].
// lane L holds column n=L%16, K=(L/16)*16+j -> 16 *contiguous* halves of row n.
// caller passes p = W + n_global*K + k_base + (L/16)*16
__device__ __forceinline__ v16h load_bT(const h16* __restrict__ p) {
  v16h b;
#pragma unroll
  for (int j = 0; j < 16; ++j) b[j] = p[j];
  return b;
}

// B operand for C = A @ V with row-major V[K][N] (tile origin src, stride ld):
// lane L holds column n=L%16 at k=(L/16)*16+j  (strided gather; used from LDS)
__device__ __forceinline__ v16h load_bN(const h16* __restrict__ src, int ld,
                                        int lm, int lhi) {
  v16h b;
  const h16* p = src + (lhi * 16) * ld + lm;
#pragma unroll
  for (int j = 0; j < 16; ++j) b[j] = p[j * ld];
  return b;
}

// ---------------------------------------------------------------------------
// f32 -> f16 conversion
// ---------------------------------------------------------------------------
__global__ void k_f32_to_f16(const float* __restrict__ in, h16* __restrict__ out,
                             size_t n) {
  size_t i = (size_t)blockIdx.x * blockDim.x + threadIdx.x;
  const size_t stride = (size_t)gridDim.x * blockDim.x;
  for (; i < n; i += stride) out[i] = (h16)in[i];
}

// ---------------------------------------------------------------------------
// Generic WMMA GEMM:  C[M,N] = A[M,K] @ W[N,K]^T (+bias)(+resid)
// one 16x16 tile per wave, grid-stride over tiles (wave-uniform -> EXEC full).
// K-loop split: branch-free main body with global_prefetch_b8 256B ahead,
// then a 4-iteration tail — keeps the load/WMMA software pipeline intact.
// ---------------------------------------------------------------------------
__global__ __launch_bounds__(256) void k_gemm_abT(
    const h16* __restrict__ A, const h16* __restrict__ W,
    const float* __restrict__ bias, const h16* __restrict__ resid,
    float* __restrict__ Cf, h16* __restrict__ Ch, int M, int N, int K) {
  const int lane = threadIdx.x & 31, lm = lane & 15, lhi = lane >> 4;
  const int wavesPerBlk = blockDim.x >> 5;
  const int gw = blockIdx.x * wavesPerBlk + (threadIdx.x >> 5);
  const int nWaves = gridDim.x * wavesPerBlk;
  const int ntn = N >> 4;
  const int tiles = (M >> 4) * ntn;
  const int kIters = K >> 5;
  const int kMain = kIters - 4;
  for (int tile = gw; tile < tiles; tile += nWaves) {
    const int mt = tile / ntn, nt = tile - mt * ntn;
    v8f acc = {};
    const h16* Ab = A + (size_t)(mt * 16) * K;
    const h16* Wb = W + (size_t)(nt * 16 + lm) * K + lhi * 16;
    int kt = 0;
    for (; kt < kMain; ++kt) {  // branch-free body + prefetch 4 k-steps ahead
      __builtin_prefetch(Wb + (kt + 4) * 32, 0, 1);
      __builtin_prefetch(Ab + lm * K + (kt + 4) * 32, 0, 1);
      v16h a = load_a16(Ab + kt * 32, K, lm, lhi);
      v16h b = load_bT(Wb + kt * 32);
      acc = wmma16(a, b, acc);
    }
    for (; kt < kIters; ++kt) {  // tail, no prefetch
      v16h a = load_a16(Ab + kt * 32, K, lm, lhi);
      v16h b = load_bT(Wb + kt * 32);
      acc = wmma16(a, b, acc);
    }
    const int ng = nt * 16 + lm;
    const float bv = bias ? bias[ng] : 0.f;
    const int mbase = mt * 16 + lhi * 8;
#pragma unroll
    for (int i = 0; i < 8; ++i) {
      const size_t off = (size_t)(mbase + i) * N + ng;
      float v = acc[i] + bv;
      if (resid) v += (float)resid[off];
      if (Cf) Cf[off] = v;
      if (Ch) Ch[off] = (h16)v;
    }
  }
}

// ---------------------------------------------------------------------------
// Sequential ON-LSTM/PFN recurrence: one persistent 32-wave workgroup.
// LDS: gates f32 [32][1280] (160KB) + c f32 [32][256] (32KB)
//    + h f16 [32][256] (16KB) + cat f16 [32][768] (48KB) = 256KB <= 320KB/WGP
// ---------------------------------------------------------------------------
__global__ __launch_bounds__(1024) void k_pfn_recurrence(
    const float* __restrict__ Gpre,  // [S,B,1280] = x@Wi^T + bi
    const h16* __restrict__ WhH,     // [1280,256]
    const float* __restrict__ bh,    // [1280]
    const h16* __restrict__ WtH,     // [256,768]
    const float* __restrict__ bt,    // [256]
    h16* __restrict__ hNer,          // [B,S,256]
    h16* __restrict__ hRe) {         // [B,S,256]
  extern __shared__ char smem[];
  float* gatesF = (float*)smem;           // 32*1280
  float* cF = gatesF + B_ * G5_;          // 32*256
  h16* hH = (h16*)(cF + B_ * H_);         // 32*256
  h16* catH = hH + B_ * H_;               // 32*768

  const int tid = threadIdx.x;
  const int wave = tid >> 5;
  const int lane = tid & 31;
  const int lm = lane & 15, lhi = lane >> 4;

  for (int i = tid; i < B_ * H_; i += 1024) { hH[i] = (h16)0.f; cF[i] = 0.f; }
  __syncthreads();

  for (int t = 0; t < S_; ++t) {
    // ---- (a) gates = Gpre[t] + h @ Wh^T + bh   (2x80 tiles, K=256) ----
    const float* gp = Gpre + (size_t)t * (B_ * G5_);
    for (int tile = wave; tile < 160; tile += 32) {
      const int mt = tile / 80, nt = tile - mt * 80;
      v8f acc = {};
      const h16* Wb = WhH + (size_t)(nt * 16 + lm) * H_ + lhi * 16;
      for (int kt = 0; kt < 8; ++kt) {
        v16h a = load_a16(hH + mt * 16 * H_ + kt * 32, H_, lm, lhi);
        v16h b = load_bT(Wb + kt * 32);
        acc = wmma16(a, b, acc);
      }
      const int ng = nt * 16 + lm;
      const float bias = bh[ng];
      const int mbase = mt * 16 + lhi * 8;
#pragma unroll
      for (int i = 0; i < 8; ++i) {
        const int m = mbase + i;
        gatesF[m * G5_ + ng] = acc[i] + gp[m * G5_ + ng] + bias;
      }
    }
    __syncthreads();
    // ---- (b) cumsoftmax in place on the 4 gate chunks (wave per row-chunk) --
    for (int task = wave; task < 128; task += 32) {
      const int b = task >> 2, ch = task & 3;
      float* row = gatesF + b * G5_ + H_ * (ch + 1);
      float v[8];
#pragma unroll
      for (int j = 0; j < 8; ++j) v[j] = row[lane * 8 + j];
      float mx = v[0];
#pragma unroll
      for (int j = 1; j < 8; ++j) mx = fmaxf(mx, v[j]);
#pragma unroll
      for (int off = 16; off > 0; off >>= 1) mx = fmaxf(mx, __shfl_xor(mx, off, 32));
      float s = 0.f;
#pragma unroll
      for (int j = 0; j < 8; ++j) { v[j] = __expf(v[j] - mx); s += v[j]; }
      float incl = s;  // inclusive scan of per-lane sums
#pragma unroll
      for (int off = 1; off < 32; off <<= 1) {
        const float o = __shfl_up(incl, off, 32);
        if (lane >= off) incl += o;
      }
      const float inv = 1.f / __shfl(incl, 31, 32);
      float run = incl - s;  // exclusive prefix
      const bool isEg = (ch & 1) == 0;  // chunks: eg_cin, rg_cin, eg_c, rg_c
#pragma unroll
      for (int j = 0; j < 8; ++j) {
        run += v[j];
        const float cs = run * inv;
        row[lane * 8 + j] = isEg ? (1.f - cs) : cs;
      }
    }
    __syncthreads();
    // ---- (c) element-wise gate mixing ----
    for (int p = tid; p < B_ * H_; p += 1024) {
      const int b = p >> 8, j = p & (H_ - 1);
      const float* gr = gatesF + b * G5_;
      const float c = tanhf(gr[j]);
      const float egi = gr[H_ + j], rgi = gr[2 * H_ + j];
      const float egc = gr[3 * H_ + j], rgc = gr[4 * H_ + j];
      const float cin = cF[p];
      const float ovc = rgc * egc, upc = rgc - ovc, dnc = egc - ovc;
      const float ovi = rgi * egi, upi = rgi - ovi, dni = egi - ovi;
      const float share = ovi * cin + ovc * c;
      const float cre = upi * cin + upc * c + share;
      const float cner = dni * cin + dnc * c + share;
      const size_t go = ((size_t)b * S_ + t) * H_ + j;  // [B,S,H] for attention
      hRe[go] = (h16)tanhf(cre);
      hNer[go] = (h16)tanhf(cner);
      h16* cr = catH + b * CAT_;
      cr[j] = (h16)cre;
      cr[H_ + j] = (h16)cner;
      cr[2 * H_ + j] = (h16)share;
    }
    __syncthreads();
    // ---- (d) c_out = cat @ Wt^T + bt ; h = tanh(c_out)  (2x16 tiles, K=768)
    {
      const int mt = wave >> 4, nt = wave & 15;
      v8f acc = {};
      const h16* Wb = WtH + (size_t)(nt * 16 + lm) * CAT_ + lhi * 16;
      for (int kt = 0; kt < 24; ++kt) {
        v16h a = load_a16(catH + mt * 16 * CAT_ + kt * 32, CAT_, lm, lhi);
        v16h b = load_bT(Wb + kt * 32);
        acc = wmma16(a, b, acc);
      }
      const int ng = nt * 16 + lm;
      const float bias = bt[ng];
      const int mbase = mt * 16 + lhi * 8;
#pragma unroll
      for (int i = 0; i < 8; ++i) {
        const float co = acc[i] + bias;
        const int m = mbase + i;
        cF[m * H_ + ng] = co;
        hH[m * H_ + ng] = (h16)tanhf(co);
      }
    }
    __syncthreads();
  }
}

// ---------------------------------------------------------------------------
// Relative multi-head attention core: one block per (b, head, 16-row q-tile).
// scores = q@k^T + q@relpos^T (K=HD=32 -> 1 WMMA per term per tile),
// row softmax in LDS, O = P@V with V staged in LDS (coalesced b128 stage-in,
// strided B-operand gathers hit LDS instead of VMEM).
// LDS: scores 32KB + partials 8KB + V slice 32KB = 72KB (<320KB/WGP)
// ---------------------------------------------------------------------------
__global__ __launch_bounds__(256) void k_attention(
    const h16* __restrict__ qH, const h16* __restrict__ kH,
    const h16* __restrict__ vH, const h16* __restrict__ relH,  // [NH,512,32]
    h16* __restrict__ oH) {
  extern __shared__ char smem[];
  float* scores = (float*)smem;        // [16][512] f32 -> probs in place
  float* pOut = scores + 16 * S_;      // [8][16][16] partials
  h16* vLds = (h16*)(pOut + 8 * 256);  // [512][32] V head slice

  const int blk = blockIdx.x;          // B*NH*(S/16) = 8192
  const int qt = blk & 31;
  const int head = (blk >> 5) & 7;
  const int b = blk >> 8;
  const int tid = threadIdx.x, wave = tid >> 5, lane = tid & 31;
  const int lm = lane & 15, lhi = lane >> 4;

  const size_t rowBase = (size_t)b * S_ * H_ + head * HD_;
  const h16* qB = qH + rowBase;
  const h16* kB = kH + rowBase;
  const h16* vB = vH + rowBase;
  const h16* rB = relH + head * (ML_ * HD_);

  // stage V head slice into LDS: 512 rows x 32 halves, 16B vector copies
  for (int idx = tid; idx < S_ * 4; idx += 256) {
    const int r = idx >> 2, c = (idx & 3) * 8;
    *(v4f*)(vLds + r * HD_ + c) = *(const v4f*)(vB + (size_t)r * H_ + c);
  }

  const v16h aq = load_a16(qB + (size_t)(qt * 16) * H_, H_, lm, lhi);

  // phase 1: scores (4 k-tiles per wave, uniform trip count)
  for (int kt = wave; kt < 32; kt += 8) {
    v8f acc = {};
    v16h bk = load_bT(kB + (size_t)(kt * 16 + lm) * H_ + lhi * 16);
    acc = wmma16(aq, bk, acc);
    v16h br = load_bT(rB + (kt * 16 + lm) * HD_ + lhi * 16);
    acc = wmma16(aq, br, acc);
#pragma unroll
    for (int i = 0; i < 8; ++i)
      scores[(lhi * 8 + i) * S_ + kt * 16 + lm] = acc[i];
  }
  __syncthreads();
  // phase 2: softmax per row (no 1/sqrt(d), per reference)
  for (int r = wave; r < 16; r += 8) {
    float* row = scores + r * S_;
    float mx = -1e30f;
    for (int j = lane; j < S_; j += 32) mx = fmaxf(mx, row[j]);
#pragma unroll
    for (int off = 16; off > 0; off >>= 1) mx = fmaxf(mx, __shfl_xor(mx, off, 32));
    float s = 0.f;
    for (int j = lane; j < S_; j += 32) {
      const float e = __expf(row[j] - mx);
      row[j] = e;
      s += e;
    }
#pragma unroll
    for (int off = 16; off > 0; off >>= 1) s += __shfl_xor(s, off, 32);
    const float inv = 1.f / s;
    for (int j = lane; j < S_; j += 32) row[j] *= inv;
  }
  __syncthreads();
  // phase 3: O = P @ V ; wave -> (n-tile = w&1, K-quarter = w>>1)
  {
    const int nt = wave & 1, kq = wave >> 1;
    v8f acc = {};
    for (int kt = kq * 4; kt < kq * 4 + 4; ++kt) {
      v16h a = load_a32(scores + kt * 32, S_, lm, lhi);
      v16h bv = load_bN(vLds + (kt * 32) * HD_ + nt * 16, HD_, lm, lhi);
      acc = wmma16(a, bv, acc);
    }
#pragma unroll
    for (int i = 0; i < 8; ++i)
      pOut[wave * 256 + (lhi * 8 + i) * 16 + lm] = acc[i];
  }
  __syncthreads();
  for (int idx = tid; idx < 16 * 32; idx += 256) {
    const int m = idx >> 5, d = idx & 31;
    const int nt = d >> 4, n = d & 15;
    float sum = 0.f;
#pragma unroll
    for (int kq = 0; kq < 4; ++kq) sum += pOut[(kq * 2 + nt) * 256 + m * 16 + n];
    oH[rowBase + (size_t)(qt * 16 + m) * H_ + d] = (h16)sum;
  }
}

// ---------------------------------------------------------------------------
// LayerNorm over H=256, one wave per row; writes transposed [S,B,H] output
// ---------------------------------------------------------------------------
__global__ __launch_bounds__(256) void k_layernorm(
    const float* __restrict__ z,  // [B,S,H] residual-added
    const float* __restrict__ g, const float* __restrict__ be,
    float* __restrict__ out) {    // [S,B,H]
  const int wave = threadIdx.x >> 5, lane = threadIdx.x & 31;
  const int row = blockIdx.x * 8 + wave;  // B*S rows
  const int b = row >> 9;                 // row / S_
  const int s = row & (S_ - 1);
  const float* zr = z + (size_t)row * H_;
  float v[8], sum = 0.f, sq = 0.f;
#pragma unroll
  for (int j = 0; j < 8; ++j) {
    v[j] = zr[lane * 8 + j];
    sum += v[j];
    sq += v[j] * v[j];
  }
#pragma unroll
  for (int off = 16; off > 0; off >>= 1) {
    sum += __shfl_xor(sum, off, 32);
    sq += __shfl_xor(sq, off, 32);
  }
  const float mean = sum * (1.f / H_);
  const float var = sq * (1.f / H_) - mean * mean;
  const float rstd = rsqrtf(var + LN_EPS);
  float* orow = out + ((size_t)s * B_ + b) * H_;
#pragma unroll
  for (int j = 0; j < 8; ++j) {
    const int col = lane * 8 + j;
    orow[col] = (v[j] - mean) * rstd * g[col] + be[col];
  }
}

// ---------------------------------------------------------------------------
extern "C" void kernel_launch(void* const* d_in, const int* in_sizes, int n_in,
                              void* d_out, int out_size, void* d_ws,
                              size_t ws_size, hipStream_t stream) {
  (void)in_sizes; (void)n_in; (void)out_size; (void)ws_size;
  const float* x = (const float*)d_in[0];
  const float* Wi = (const float*)d_in[1];
  const float* bi = (const float*)d_in[2];
  const float* Wh = (const float*)d_in[3];
  const float* bh = (const float*)d_in[4];
  const float* Wt = (const float*)d_in[5];
  const float* bt = (const float*)d_in[6];
  const float* Wq = (const float*)d_in[7];
  const float* bq = (const float*)d_in[8];
  const float* Wk = (const float*)d_in[9];
  const float* bk = (const float*)d_in[10];
  const float* Wv = (const float*)d_in[11];
  const float* bv = (const float*)d_in[12];
  const float* Wo = (const float*)d_in[13];
  const float* bo = (const float*)d_in[14];
  const float* relpos = (const float*)d_in[15];
  const float* ln_g = (const float*)d_in[16];
  const float* ln_b = (const float*)d_in[17];

  size_t off = 0;
  auto alloc = [&](size_t bytes) -> void* {
    void* p = (char*)d_ws + off;
    off += (bytes + 255) & ~(size_t)255;
    return p;
  };
  h16* xH = (h16*)alloc((size_t)SB_ * I_ * 2);
  h16* WiH = (h16*)alloc((size_t)G5_ * I_ * 2);
  h16* WhH = (h16*)alloc((size_t)G5_ * H_ * 2);
  h16* WtH = (h16*)alloc((size_t)H_ * CAT_ * 2);
  h16* WqH = (h16*)alloc((size_t)3 * H_ * H_ * 2);
  h16* WkH = (h16*)alloc((size_t)3 * H_ * H_ * 2);
  h16* WvH = (h16*)alloc((size_t)3 * H_ * H_ * 2);
  h16* WoH = (h16*)alloc((size_t)3 * H_ * H_ * 2);
  h16* relH = (h16*)alloc((size_t)3 * NH_ * ML_ * HD_ * 2);
  float* GpreF = (float*)alloc((size_t)SB_ * G5_ * 4);
  h16* hNerH = (h16*)alloc((size_t)SB_ * H_ * 2);
  h16* hReH = (h16*)alloc((size_t)SB_ * H_ * 2);
  h16* qBuf = (h16*)alloc((size_t)SB_ * H_ * 2);
  h16* kBuf = (h16*)alloc((size_t)SB_ * H_ * 2);
  h16* vBuf = (h16*)alloc((size_t)SB_ * H_ * 2);
  h16* oBuf = (h16*)alloc((size_t)SB_ * H_ * 2);
  float* zF = (float*)alloc((size_t)SB_ * H_ * 4);

  auto cvt = [&](const float* src, h16* dst, size_t n) {
    const int blocks = (int)((n + 2047) / 2048);
    k_f32_to_f16<<<blocks, 256, 0, stream>>>(src, dst, n);
  };
  cvt(x, xH, (size_t)SB_ * I_);
  cvt(Wi, WiH, (size_t)G5_ * I_);
  cvt(Wh, WhH, (size_t)G5_ * H_);
  cvt(Wt, WtH, (size_t)H_ * CAT_);
  cvt(Wq, WqH, (size_t)3 * H_ * H_);
  cvt(Wk, WkH, (size_t)3 * H_ * H_);
  cvt(Wv, WvH, (size_t)3 * H_ * H_);
  cvt(Wo, WoH, (size_t)3 * H_ * H_);
  cvt(relpos, relH, (size_t)3 * NH_ * ML_ * HD_);

  // Gpre = x @ Wi^T + bi : [16384,1280], K=768
  {
    const int tiles = (SB_ / 16) * (G5_ / 16);
    k_gemm_abT<<<(tiles + 7) / 8, 256, 0, stream>>>(
        xH, WiH, bi, nullptr, GpreF, nullptr, SB_, G5_, I_);
  }
  // sequential recurrence: one 32-wave workgroup, 256KB dynamic LDS
  {
    const size_t smem = (size_t)B_ * G5_ * 4 + (size_t)B_ * H_ * 4 +
                        (size_t)B_ * H_ * 2 + (size_t)B_ * CAT_ * 2;
    (void)hipFuncSetAttribute((const void*)k_pfn_recurrence,
                              hipFuncAttributeMaxDynamicSharedMemorySize,
                              (int)smem);
    k_pfn_recurrence<<<1, 1024, smem, stream>>>(GpreF, WhH, bh, WtH, bt, hNerH,
                                                hReH);
  }
  // three attention passes: i=0 on h_ner, i=1 on h_re, i=2 on h_re (faithful)
  const size_t smemA =
      (size_t)16 * S_ * 4 + (size_t)8 * 256 * 4 + (size_t)S_ * HD_ * 2;  // 72KB
  (void)hipFuncSetAttribute((const void*)k_attention,
                            hipFuncAttributeMaxDynamicSharedMemorySize,
                            (int)smemA);
  for (int i = 0; i < 3; ++i) {
    const h16* src = (i == 0) ? hNerH : hReH;
    const int tiles = (SB_ / 16) * (H_ / 16);
    const int gblk = (tiles + 7) / 8;
    k_gemm_abT<<<gblk, 256, 0, stream>>>(src, WqH + (size_t)i * H_ * H_,
                                         bq + i * H_, nullptr, nullptr, qBuf,
                                         SB_, H_, H_);
    k_gemm_abT<<<gblk, 256, 0, stream>>>(src, WkH + (size_t)i * H_ * H_,
                                         bk + i * H_, nullptr, nullptr, kBuf,
                                         SB_, H_, H_);
    k_gemm_abT<<<gblk, 256, 0, stream>>>(src, WvH + (size_t)i * H_ * H_,
                                         bv + i * H_, nullptr, nullptr, vBuf,
                                         SB_, H_, H_);
    k_attention<<<B_ * NH_ * (S_ / 16), 256, smemA, stream>>>(
        qBuf, kBuf, vBuf, relH + (size_t)i * NH_ * ML_ * HD_, oBuf);
    // z = attnO @ Wo^T + bo + residual(src)
    k_gemm_abT<<<gblk, 256, 0, stream>>>(oBuf, WoH + (size_t)i * H_ * H_,
                                         bo + i * H_, src, zF, nullptr, SB_, H_,
                                         H_);
    k_layernorm<<<(B_ * S_) / 8, 256, 0, stream>>>(
        zF, ln_g, ln_b, (float*)d_out + (size_t)i * S_ * B_ * H_);
  }
}
